// MulModel_Four_modules_LLM_65249143161444
// MI455X (gfx1250) — matrix operand order repaired and checked by reading, not compiled, not run on hardware
//
#include <hip/hip_runtime.h>
#include <hip/hip_bf16.h>
#include <math.h>

// ---------------------------------------------------------------------------
// MI455X (gfx1250) implementation.
//   * all GEMMs: v_wmma_f32_16x16x32_bf16, 128x128x64 tile, 8 wave32/block,
//     each wave owns a 2x4 grid of 16x16 accumulators (16 wmma / K-tile).
//   * A/W operands stored bf16 in HBM (halved read traffic, fp32 accum).
//   * Double-buffered LDS: A tile staged by TDM tensor_load_to_lds
//     (TENSORcnt), B tile staged+transposed via VGPRs, both for tile i+1
//     issued before computing tile i; one barrier per K-tile.
// ---------------------------------------------------------------------------

typedef __attribute__((ext_vector_type(16))) __bf16 v16bf;
typedef __attribute__((ext_vector_type(8)))  float  v8f;
typedef unsigned int uint32x4 __attribute__((ext_vector_type(4)));
typedef int          int32x4  __attribute__((ext_vector_type(4)));
typedef int          int32x8  __attribute__((ext_vector_type(8)));

#ifndef __has_builtin
#define __has_builtin(x) 0
#endif
#if __has_builtin(__builtin_amdgcn_tensor_load_to_lds) && \
    __has_builtin(__builtin_amdgcn_s_wait_tensorcnt)
#define USE_TDM 1
#else
#define USE_TDM 0
#endif

#define TILE_M 128
#define TILE_N 128
#define TILE_K 64
#define LDSW_A 64   // A-tile LDS row stride (ushorts)
#define LDSW_B 72   // B-tile LDS row stride (64 data + 8 pad, 144B = 16B-aligned)

static __device__ __forceinline__ unsigned short f2bf(float f) {
    unsigned int u = __float_as_uint(f);
    u += 0x7FFFu + ((u >> 16) & 1u);          // round-to-nearest-even
    return (unsigned short)(u >> 16);
}
static __device__ __forceinline__ float bf2f(unsigned short h) {
    return __uint_as_float(((unsigned int)h) << 16);
}

union FragAB { v16bf v; uint4 q[2]; };
union FragC  { v8f  v; float f[8]; };

// C[rows x N](f32, ldc) = A[rows x K](bf16, lda) * W[K x N](bf16, row-major) + bias
// Requires rows%128==0, N%128==0, K%64==0.
__global__ __launch_bounds__(256) void gemm_bf16_wmma_kernel(
    const unsigned short* __restrict__ A, int lda,
    const unsigned short* __restrict__ W, int N,
    const float* __restrict__ bias,
    float* __restrict__ C, int ldc,
    int K)
{
    __shared__ __align__(16) unsigned short lA[2][TILE_M * LDSW_A];  // 2 x 16KB
    __shared__ __align__(16) unsigned short lB[2][TILE_N * LDSW_B];  // 2 x 18KB

    const int t     = threadIdx.x;
    const int lane  = t & 31;
    const int wave  = t >> 5;
    const int waveM = wave & 3;    // 4 waves along M
    const int waveN = wave >> 2;   // 2 waves along N
    const long rowBase = (long)blockIdx.y * TILE_M;
    const int  colBase = blockIdx.x * TILE_N;

    FragC acc[2][4];
#pragma unroll
    for (int mi = 0; mi < 2; ++mi)
#pragma unroll
        for (int ni = 0; ni < 4; ++ni)
#pragma unroll
            for (int i = 0; i < 8; ++i) acc[mi][ni].f[i] = 0.0f;

    // staging coordinates
    const int bK16 = t >> 4;         // 0..15 (+16,+32,+48)
    const int bC8  = (t & 15) * 8;   // 0..120
#if !USE_TDM
    const int aR = t >> 1;           // 0..127
    const int aC = (t & 1) * 8;      // 0 or 8 (+16,+32,+48)
#endif

    // fragment coordinates (ISA 7.12.2 layouts)
    const int mrow  = lane & 15;
    const int kHalf = lane >> 4;

#if USE_TDM
    const int rowsTotal = (int)gridDim.y * TILE_M;
    // lA is written only by the TDM engine; keep a (never-taken, but not
    // provably dead) store so alias analysis keeps the LDS reads alive.
    if (K == 0) { lA[0][t] = (unsigned short)N; lA[1][t] = (unsigned short)N; }
#endif

    // stage K-tile starting at k0 into LDS buffer `buf`
    auto stage = [&](int k0, int buf) {
#if USE_TDM
        if (wave == 0) {
            // D# for a 2D tile: 128 rows x 64 cols of 2-byte elements.
            const unsigned ldsA = (unsigned)(unsigned long)(void*)&lA[buf][0];
            const unsigned long long ga =
                (unsigned long long)(unsigned long)A +
                2ull * ((unsigned long long)rowBase * (unsigned)lda + (unsigned)k0);
            uint32x4 g0;
            g0[0] = 1u;                                   // count=1 (valid)
            g0[1] = ldsA;                                 // lds_addr
            g0[2] = (unsigned)ga;                         // global_addr[31:0]
            g0[3] = (unsigned)((ga >> 32) & 0x01ffffffull) | (2u << 30); // [56:32]+type=2
            int32x8 g1;
            g1[0] = 1 << 16;                              // data_size=1 (2 bytes)
            g1[1] = (int)(((unsigned)K & 0xffffu) << 16); // tensor_dim0[15:0]
            g1[2] = (int)((unsigned)K >> 16)              // tensor_dim0[31:16]
                  | (int)(((unsigned)rowsTotal & 0xffffu) << 16); // tensor_dim1[15:0]
            g1[3] = (int)(((unsigned)rowsTotal >> 16) & 0xffffu)  // tensor_dim1[31:16]
                  | (int)((unsigned)TILE_K << 16);        // tile_dim0 = 64
            g1[4] = TILE_M;                               // tile_dim1 = 128
            g1[5] = lda;                                  // tensor_dim0_stride[31:0]
            g1[6] = 0;
            g1[7] = 0;
            int32x4 gz = {0, 0, 0, 0};
#if defined(__clang_major__) && __clang_major__ >= 23
            int32x8 gz8 = {0, 0, 0, 0, 0, 0, 0, 0};
            __builtin_amdgcn_tensor_load_to_lds(g0, g1, gz, gz, gz8, 0);
#else
            __builtin_amdgcn_tensor_load_to_lds(g0, g1, gz, gz, 0);
#endif
        }
#else
        // Fallback: VGPR staging of the bf16 A tile (128x64).
#pragma unroll
        for (int i = 0; i < 4; ++i) {
            const int c = aC + 16 * i;
            *(uint4*)&lA[buf][aR * LDSW_A + c] =
                *(const uint4*)(A + (rowBase + aR) * (long)lda + k0 + c);
        }
#endif
        // stage B tile (64x128 bf16) transposed into LDS as [N][K]
#pragma unroll
        for (int i = 0; i < 4; ++i) {
            const int kk = bK16 + 16 * i;
            union { uint4 q; unsigned short u[8]; } w;
            w.q = *(const uint4*)(W + (long)(k0 + kk) * N + colBase + bC8);
#pragma unroll
            for (int j = 0; j < 8; ++j) lB[buf][(bC8 + j) * LDSW_B + kk] = w.u[j];
        }
        if (k0 + 2 * TILE_K < K)  // prefetch B two tiles ahead (global_prefetch_b8)
            __builtin_prefetch(W + (long)(k0 + 2 * TILE_K + bK16) * N + colBase + bC8, 0, 1);
    };

    const int nTiles = K / TILE_K;

    // prologue: stage tile 0
    stage(0, 0);
#if USE_TDM
    if (wave == 0) __builtin_amdgcn_s_wait_tensorcnt(0);
#endif
    __syncthreads();

    for (int ti = 0; ti < nTiles; ++ti) {
        const int cur = ti & 1;
        const bool more = (ti + 1) < nTiles;
        if (more) stage((ti + 1) * TILE_K, cur ^ 1);  // overlap with compute

        // compute 16 wmma from buffer `cur` (two K=32 chunks)
        const unsigned short* __restrict__ cA = &lA[cur][0];
        const unsigned short* __restrict__ cB = &lB[cur][0];
#pragma unroll
        for (int kc = 0; kc < 2; ++kc) {
            FragAB a[2], b[4];
#pragma unroll
            for (int mi = 0; mi < 2; ++mi) {
                const int m = waveM * 32 + mi * 16 + mrow;
                const uint4* p = (const uint4*)&cA[m * LDSW_A + kc * 32];
                a[mi].q[0] = p[kHalf];       // K = kHalf*8 .. +7
                a[mi].q[1] = p[2 + kHalf];   // K = 16 + kHalf*8 .. +7
            }
#pragma unroll
            for (int ni = 0; ni < 4; ++ni) {
                const int n = waveN * 64 + ni * 16 + mrow;
                const uint4* p = (const uint4*)&cB[n * LDSW_B + kc * 32];
                b[ni].q[0] = p[kHalf];
                b[ni].q[1] = p[2 + kHalf];
            }
#pragma unroll
            for (int mi = 0; mi < 2; ++mi)
#pragma unroll
                for (int ni = 0; ni < 4; ++ni)
                    acc[mi][ni].v = __builtin_amdgcn_wmma_f32_16x16x32_bf16(
                        false, a[mi].v, false, b[ni].v,
                        (short)0, acc[mi][ni].v, false, false);
        }

#if USE_TDM
        if (more && wave == 0) __builtin_amdgcn_s_wait_tensorcnt(0);
#endif
        __syncthreads();
    }

    // epilogue: one 64-bit base, 32-bit offsets. C/D layout: VGPR i ->
    // M = i + 8*(lane>=16), N = lane&15.
    float* __restrict__ Cb = C + rowBase * (long)ldc + colBase;
    const int hi8 = kHalf * 8;
#pragma unroll
    for (int ni = 0; ni < 4; ++ni) {
        const int colOff = waveN * 64 + ni * 16 + mrow;
        const float bv = bias ? bias[colBase + colOff] : 0.0f;
#pragma unroll
        for (int mi = 0; mi < 2; ++mi) {
            const int rowOff = waveM * 32 + mi * 16 + hi8;
#pragma unroll
            for (int i = 0; i < 8; ++i)
                Cb[(rowOff + i) * ldc + colOff] = acc[mi][ni].f[i] + bv;
        }
    }
}

// fp32 -> bf16 bulk conversion
__global__ void f32_to_bf16_kernel(const float* __restrict__ X,
                                   unsigned short* __restrict__ Y, long n)
{
    for (long i = (long)blockIdx.x * blockDim.x + threadIdx.x; i < n;
         i += (long)gridDim.x * blockDim.x)
        Y[i] = f2bf(X[i]);
}

// Per-column mean/var (biased) over rows of contiguous fp32 [rows x cols].
__global__ __launch_bounds__(256) void col_stats_kernel(
    const float* __restrict__ X, long rows, int cols,
    float* __restrict__ mean, float* __restrict__ var)
{
    const int cx = blockIdx.x * 64 + (threadIdx.x & 63);
    const int ry = threadIdx.x >> 6;  // 0..3
    float s = 0.0f, ss = 0.0f;
    for (long r = ry; r < rows; r += 4) {
        const float v = X[r * (long)cols + cx];
        s += v; ss += v * v;
    }
    __shared__ float sm[4][64];
    __shared__ float sv[4][64];
    sm[ry][threadIdx.x & 63] = s;
    sv[ry][threadIdx.x & 63] = ss;
    __syncthreads();
    if (ry == 0) {
        const int x = threadIdx.x & 63;
        s  = sm[0][x] + sm[1][x] + sm[2][x] + sm[3][x];
        ss = sv[0][x] + sv[1][x] + sv[2][x] + sv[3][x];
        const float m = s / (float)rows;
        mean[cx] = m;
        var[cx]  = ss / (float)rows - m * m;
    }
}

// y = bn(x)*g+be (+leaky_relu); output fp32 or bf16
__global__ void bn_act_kernel(
    const float* __restrict__ X, void* __restrict__ Y,
    const float* __restrict__ mean, const float* __restrict__ var,
    const float* __restrict__ g, const float* __restrict__ be,
    long n, int cols, int do_lrelu, int out_bf16)
{
    for (long i = (long)blockIdx.x * blockDim.x + threadIdx.x; i < n;
         i += (long)gridDim.x * blockDim.x) {
        const int c = (int)(i % cols);
        const float x = X[i];
        float y = (x - mean[c]) * rsqrtf(var[c] + 1e-5f) * g[c] + be[c];
        if (do_lrelu) y = (y > 0.0f) ? y : 0.01f * y;
        if (out_bf16) ((unsigned short*)Y)[i] = f2bf(y);
        else          ((float*)Y)[i] = y;
    }
}

// fused = lrelu(bn(aF)) * sigmoid(bn(gate)) -> bf16
__global__ void fuse_gate_kernel(
    const float* __restrict__ aF, const float* __restrict__ gate,
    const float* __restrict__ mF, const float* __restrict__ vF,
    const float* __restrict__ gF, const float* __restrict__ bF,
    const float* __restrict__ mG, const float* __restrict__ vG,
    const float* __restrict__ gG, const float* __restrict__ bG,
    unsigned short* __restrict__ out, long n, int cols)
{
    for (long i = (long)blockIdx.x * blockDim.x + threadIdx.x; i < n;
         i += (long)gridDim.x * blockDim.x) {
        const int c = (int)(i % cols);
        float a = (aF[i] - mF[c]) * rsqrtf(vF[c] + 1e-5f) * gF[c] + bF[c];
        a = (a > 0.0f) ? a : 0.01f * a;
        const float q = (gate[i] - mG[c]) * rsqrtf(vG[c] + 1e-5f) * gG[c] + bG[c];
        const float sg = 1.0f / (1.0f + __expf(-q));
        out[i] = f2bf(a * sg);
    }
}

// X[b*4+m][k] (bf16) from a/c/d (bf16) and t (fp32)
__global__ void build_x_kernel(
    const unsigned short* __restrict__ a, const unsigned short* __restrict__ c,
    const unsigned short* __restrict__ d, const float* __restrict__ t,
    unsigned short* __restrict__ X, long n)
{
    for (long i = (long)blockIdx.x * blockDim.x + threadIdx.x; i < n;
         i += (long)gridDim.x * blockDim.x) {
        const long b = i >> 10;
        const int rem = (int)(i & 1023);
        const int m = rem >> 8;
        const int k = rem & 255;
        float val;
        if (m == 3) val = t[b * 256 + k];
        else {
            const unsigned short* s = (m == 0) ? a : ((m == 1) ? c : d);
            val = bf2f(s[b * 256 + k]);
        }
        X[i] = f2bf(val);
    }
}

// 4-token, 2-head attention: one wave32 per (b,h); lanes split hd=128 into 4.
__global__ __launch_bounds__(256) void attn_kernel(
    const float* __restrict__ qkv, unsigned short* __restrict__ O, long nbh)
{
    const long w = (long)blockIdx.x * 8 + (threadIdx.x >> 5);
    if (w >= nbh) return;
    const long b = w >> 1;
    const int  hh = (int)(w & 1);
    const int  lane = threadIdx.x & 31;

    float4 q[4], k[4], v[4];
#pragma unroll
    for (int s = 0; s < 4; ++s) {
        const float* base = qkv + ((b * 4 + s) * 768 + hh * 128 + lane * 4);
        q[s] = *(const float4*)(base);
        k[s] = *(const float4*)(base + 256);
        v[s] = *(const float4*)(base + 512);
    }
    const float scale = 0.0883883476483184f;  // 1/sqrt(128)
#pragma unroll
    for (int s = 0; s < 4; ++s) {
        float sc[4];
#pragma unroll
        for (int tt = 0; tt < 4; ++tt) {
            float p = q[s].x * k[tt].x + q[s].y * k[tt].y +
                      q[s].z * k[tt].z + q[s].w * k[tt].w;
#pragma unroll
            for (int off = 16; off > 0; off >>= 1) p += __shfl_xor(p, off, 32);
            sc[tt] = p * scale;
        }
        const float mx = fmaxf(fmaxf(sc[0], sc[1]), fmaxf(sc[2], sc[3]));
        const float e0 = __expf(sc[0] - mx), e1 = __expf(sc[1] - mx);
        const float e2 = __expf(sc[2] - mx), e3 = __expf(sc[3] - mx);
        const float inv = 1.0f / (e0 + e1 + e2 + e3);
        ushort4 o;
        o.x = f2bf((e0 * v[0].x + e1 * v[1].x + e2 * v[2].x + e3 * v[3].x) * inv);
        o.y = f2bf((e0 * v[0].y + e1 * v[1].y + e2 * v[2].y + e3 * v[3].y) * inv);
        o.z = f2bf((e0 * v[0].z + e1 * v[1].z + e2 * v[2].z + e3 * v[3].z) * inv);
        o.w = f2bf((e0 * v[0].w + e1 * v[1].w + e2 * v[2].w + e3 * v[3].w) * inv);
        *(ushort4*)(O + ((b * 4 + s) * 256 + hh * 128 + lane * 4)) = o;
    }
}

// out[b, 0:2] = h2[b,:] @ cw[256,2] + cb  (one wave per row)
__global__ __launch_bounds__(256) void head_kernel(
    const float* __restrict__ H, const float* __restrict__ cw,
    const float* __restrict__ cb, float* __restrict__ out, long nrows)
{
    const long row = (long)blockIdx.x * 8 + (threadIdx.x >> 5);
    if (row >= nrows) return;
    const int lane = threadIdx.x & 31;
    float s0 = 0.0f, s1 = 0.0f;
#pragma unroll
    for (int i = 0; i < 8; ++i) {
        const int k = lane + 32 * i;
        const float h = H[row * 256 + k];
        s0 += h * cw[2 * k];
        s1 += h * cw[2 * k + 1];
    }
#pragma unroll
    for (int off = 16; off > 0; off >>= 1) {
        s0 += __shfl_xor(s0, off, 32);
        s1 += __shfl_xor(s1, off, 32);
    }
    if (lane == 0) {
        out[row * 2]     = s0 + cb[0];
        out[row * 2 + 1] = s1 + cb[1];
    }
}

// ---------------------------------------------------------------------------

extern "C" void kernel_launch(void* const* d_in, const int* in_sizes, int n_in,
                              void* d_out, int out_size, void* d_ws, size_t ws_size,
                              hipStream_t stream)
{
    (void)in_sizes; (void)n_in; (void)out_size; (void)ws_size;
    const long Bsz = 16384;
    const int  D = 256, MAP = 1024, E = 1024, M = 4;

    int idx = 0;
    auto nf = [&]() { return (const float*)d_in[idx++]; };
    const float* ast  = nf();
    const float* cfg  = nf();
    const float* pdg  = nf();
    const float* text = nf();
    const float* llm_w = nf();
    const float* llm_b = nf();
    struct BP { const float *w1, *b1, *g1, *be1, *w2, *b2, *g2, *be2; } bp[3];
    for (int i = 0; i < 3; ++i) {
        bp[i].w1 = nf(); bp[i].b1 = nf(); bp[i].g1 = nf(); bp[i].be1 = nf();
        bp[i].w2 = nf(); bp[i].b2 = nf(); bp[i].g2 = nf(); bp[i].be2 = nf();
    }
    const float* wqkv = nf();
    const float* wo   = nf();
    const float* bo   = nf();
    const float* fw  = nf(); const float* fb  = nf(); const float* fg  = nf(); const float* fbe = nf();
    const float* aw  = nf(); const float* ab  = nf(); const float* ag  = nf(); const float* abe = nf();
    const float* mw1 = nf(); const float* mb1 = nf(); const float* mg1 = nf(); const float* mbe1 = nf();
    const float* mw2 = nf(); const float* mb2 = nf(); const float* mg2 = nf(); const float* mbe2 = nf();
    const float* cw = nf();
    const float* cb = nf();

    char* ws = (char*)d_ws;
    size_t off = 0;
    auto allocB = [&](size_t bytes) {
        void* p = ws + off;
        off += (bytes + 255) & ~(size_t)255;
        return p;
    };
    auto allocF = [&](size_t n) { return (float*)allocB(n * 4); };
    auto allocH = [&](size_t n) { return (unsigned short*)allocB(n * 2); };

    // bf16 copies of GEMM operands
    unsigned short* tex16 = allocH((size_t)Bsz * E);
    unsigned short* in16[3]; for (int i = 0; i < 3; ++i) in16[i] = allocH((size_t)Bsz * D);
    unsigned short* llmw16 = allocH((size_t)E * D);
    unsigned short* w116[3], *w216[3];
    for (int i = 0; i < 3; ++i) { w116[i] = allocH((size_t)D * MAP); w216[i] = allocH((size_t)MAP * D); }
    unsigned short* wqkv16 = allocH((size_t)D * 3 * D);
    unsigned short* wo16   = allocH((size_t)D * D);
    unsigned short* fw16   = allocH((size_t)M * D * MAP);
    unsigned short* aw16   = allocH((size_t)M * M * D * MAP);
    unsigned short* mw116  = allocH((size_t)M * MAP * MAP);
    unsigned short* mw216  = allocH((size_t)MAP * D);

    // activations
    float*          t0     = allocF((size_t)Bsz * D);
    float*          brPre  = allocF((size_t)Bsz * MAP);
    unsigned short* brH16  = allocH((size_t)Bsz * MAP);
    float*          brPre2 = allocF((size_t)Bsz * D);
    unsigned short* outb16[3]; for (int i = 0; i < 3; ++i) outb16[i] = allocH((size_t)Bsz * D);
    unsigned short* X16    = allocH((size_t)Bsz * M * D);
    float*          qkv    = allocF((size_t)Bsz * M * 3 * D);
    unsigned short* O16    = allocH((size_t)Bsz * M * D);
    float*          O2     = allocF((size_t)Bsz * M * D);
    unsigned short* O2b    = allocH((size_t)Bsz * M * D);
    float*          aFp    = allocF((size_t)Bsz * M * MAP);
    float*          gateP  = allocF((size_t)Bsz * M * MAP);
    unsigned short* fus16  = allocH((size_t)Bsz * M * MAP);
    float*          hPre   = allocF((size_t)Bsz * MAP);
    unsigned short* h116   = allocH((size_t)Bsz * MAP);
    float*          h2Pre  = allocF((size_t)Bsz * D);
    float*          h2     = allocF((size_t)Bsz * D);
    float* mean1 = allocF(4096); float* var1 = allocF(4096);
    float* mean2 = allocF(4096); float* var2 = allocF(4096);

    auto conv = [&](const float* src, unsigned short* dst, long n) {
        f32_to_bf16_kernel<<<2048, 256, 0, stream>>>(src, dst, n);
    };
    auto gemm = [&](const unsigned short* A, int lda, const unsigned short* W, int N,
                    const float* bias, float* C, int ldc, int K, long rows) {
        dim3 grid((unsigned)(N / TILE_N), (unsigned)(rows / TILE_M));
        gemm_bf16_wmma_kernel<<<grid, 256, 0, stream>>>(A, lda, W, N, bias, C, ldc, K);
    };
    auto stats = [&](const float* Xp, int cols, float* mean, float* var, long rows) {
        col_stats_kernel<<<cols / 64, 256, 0, stream>>>(Xp, rows, cols, mean, var);
    };
    auto bnact = [&](const float* Xp, void* Y, const float* mean, const float* var,
                     const float* g, const float* be, long rows, int cols,
                     int act, int obf) {
        bn_act_kernel<<<8192, 256, 0, stream>>>(Xp, Y, mean, var, g, be,
                                                rows * (long)cols, cols, act, obf);
    };

    // 0) one-time bf16 conversions (weights + network inputs)
    conv(text, tex16, (long)Bsz * E);
    conv(ast, in16[0], (long)Bsz * D);
    conv(cfg, in16[1], (long)Bsz * D);
    conv(pdg, in16[2], (long)Bsz * D);
    conv(llm_w, llmw16, (long)E * D);
    for (int i = 0; i < 3; ++i) {
        conv(bp[i].w1, w116[i], (long)D * MAP);
        conv(bp[i].w2, w216[i], (long)MAP * D);
    }
    conv(wqkv, wqkv16, (long)D * 3 * D);
    conv(wo,   wo16,   (long)D * D);
    conv(fw,   fw16,   (long)M * D * MAP);
    conv(aw,   aw16,   (long)M * M * D * MAP);
    conv(mw1,  mw116,  (long)M * MAP * MAP);
    conv(mw2,  mw216,  (long)MAP * D);

    // 1) t = text @ llm_w + llm_b
    gemm(tex16, E, llmw16, D, llm_b, t0, D, E, Bsz);

    // 2) three branch MLPs
    for (int i = 0; i < 3; ++i) {
        gemm(in16[i], D, w116[i], MAP, bp[i].b1, brPre, MAP, D, Bsz);
        stats(brPre, MAP, mean1, var1, Bsz);
        bnact(brPre, brH16, mean1, var1, bp[i].g1, bp[i].be1, Bsz, MAP, 1, 1);
        gemm(brH16, MAP, w216[i], D, bp[i].b2, brPre2, D, MAP, Bsz);
        stats(brPre2, D, mean1, var1, Bsz);
        bnact(brPre2, outb16[i], mean1, var1, bp[i].g2, bp[i].be2, Bsz, D, 1, 1);
    }

    // 3) X = stack([a, c, d, t], axis=1)
    build_x_kernel<<<8192, 256, 0, stream>>>(outb16[0], outb16[1], outb16[2],
                                             t0, X16, (long)Bsz * M * D);

    // 4) qkv = X @ wqkv (no bias)
    gemm(X16, D, wqkv16, 3 * D, nullptr, qkv, 3 * D, D, Bsz * M);

    // 5) attention
    attn_kernel<<<(unsigned)(Bsz * 2 / 8), 256, 0, stream>>>(qkv, O16, Bsz * 2);

    // 6) o = o @ wo + bo ; all_emb = o.reshape(B, M*D)
    gemm(O16, D, wo16, D, bo, O2, D, D, Bsz * M);
    conv(O2, O2b, (long)Bsz * M * D);

    // 7) aF_pre (strided-A) and 8) gate_pre
    for (int m = 0; m < M; ++m) {
        gemm(O2b + m * D, M * D, fw16 + (size_t)m * D * MAP, MAP,
             fb + m * MAP, aFp + m * MAP, M * MAP, D, Bsz);
        gemm(O2b, M * D, aw16 + (size_t)m * M * D * MAP, MAP,
             ab + m * MAP, gateP + m * MAP, M * MAP, M * D, Bsz);
    }
    stats(aFp, M * MAP, mean1, var1, Bsz);
    stats(gateP, M * MAP, mean2, var2, Bsz);
    fuse_gate_kernel<<<8192, 256, 0, stream>>>(aFp, gateP,
                                               mean1, var1, fg, fbe,
                                               mean2, var2, ag, abe,
                                               fus16, (long)Bsz * M * MAP, M * MAP);

    // 9) final MLP
    gemm(fus16, M * MAP, mw116, MAP, mb1, hPre, MAP, M * MAP, Bsz);
    stats(hPre, MAP, mean1, var1, Bsz);
    bnact(hPre, h116, mean1, var1, mg1, mbe1, Bsz, MAP, 1, 1);
    gemm(h116, MAP, mw216, D, mb2, h2Pre, D, MAP, Bsz);
    stats(h2Pre, D, mean1, var1, Bsz);
    bnact(h2Pre, h2, mean1, var1, mg2, mbe2, Bsz, D, 1, 0);

    // 10) classifier head
    head_kernel<<<(unsigned)(Bsz / 8), 256, 0, stream>>>(h2, cw, cb,
                                                         (float*)d_out, Bsz);
}
